// OptimizedMoELayer_57655640982113
// MI455X (gfx1250) — matrix operand order repaired and checked by reading, not compiled, not run on hardware
//
#include <hip/hip_runtime.h>
#include <hip/hip_bf16.h>

typedef __attribute__((ext_vector_type(16))) __bf16 v16bf;
typedef __attribute__((ext_vector_type(8)))  __bf16 v8bf;
typedef __attribute__((ext_vector_type(8)))  float  v8f;

#define T_TOK 4096
#define D_DIM 1024
#define E_EXP 8
#define H_DIM 1792

// ---------------- init: zero output & workspace ----------------
__global__ __launch_bounds__(256) void moe_init(float* __restrict__ out,
                                                int* __restrict__ cnt,
                                                float* __restrict__ mean,
                                                int* __restrict__ toks,
                                                float* __restrict__ wts) {
  int i = blockIdx.x * 256 + threadIdx.x;
  if (i < T_TOK * D_DIM) out[i] = 0.f;
  if (i < E_EXP) { cnt[i] = 0; mean[i] = 0.f; }
  if (i < E_EXP * T_TOK) { toks[i] = 0; wts[i] = 0.f; }
}

// ---------------- cast x to bf16 once (used by expert GEMMs) ----------------
__global__ __launch_bounds__(256) void moe_cast(const float* __restrict__ x,
                                                __bf16* __restrict__ xbf) {
  int i = (blockIdx.x * 256 + threadIdx.x) * 8;
  float4 v0 = *(const float4*)(x + i);
  float4 v1 = *(const float4*)(x + i + 4);
  v8bf o;
  o[0] = (__bf16)v0.x; o[1] = (__bf16)v0.y; o[2] = (__bf16)v0.z; o[3] = (__bf16)v0.w;
  o[4] = (__bf16)v1.x; o[5] = (__bf16)v1.y; o[6] = (__bf16)v1.z; o[7] = (__bf16)v1.w;
  *(v8bf*)(xbf + i) = o;
}

// ---------------- gate: logits (bf16), top2 softmax, scatter ----------------
__global__ __launch_bounds__(256) void moe_gate(const float* __restrict__ x,
                                                const float* __restrict__ gw,
                                                int* __restrict__ cnt,
                                                float* __restrict__ mean,
                                                int* __restrict__ toks,
                                                float* __restrict__ wts) {
  int lane = threadIdx.x & 31;
  int wid  = threadIdx.x >> 5;
  int t    = blockIdx.x * 8 + wid;

  float acc[E_EXP];
#pragma unroll
  for (int e = 0; e < E_EXP; e++) acc[e] = 0.f;

  const float* xr = x + (size_t)t * D_DIM;
  for (int i = 0; i < D_DIM / 32; i++) {
    int d = lane + i * 32;
    float xv = (float)(__bf16)xr[d];
#pragma unroll
    for (int e = 0; e < E_EXP; e++)
      acc[e] += xv * (float)(__bf16)gw[e * D_DIM + d];
  }
#pragma unroll
  for (int e = 0; e < E_EXP; e++) {
#pragma unroll
    for (int off = 16; off > 0; off >>= 1)
      acc[e] += __shfl_xor(acc[e], off, 32);
  }

  if (lane == 0) {
    float lg[E_EXP];
#pragma unroll
    for (int e = 0; e < E_EXP; e++) lg[e] = acc[e];
    int i1 = 0; float l1 = lg[0];
#pragma unroll
    for (int e = 1; e < E_EXP; e++) if (lg[e] > l1) { l1 = lg[e]; i1 = e; }
    int i2 = -1; float l2 = -3.4e38f;
#pragma unroll
    for (int e = 0; e < E_EXP; e++) if (e != i1 && lg[e] > l2) { l2 = lg[e]; i2 = e; }
    float ex = __expf(l2 - l1);
    float w1 = 1.f / (1.f + ex);
    float w2 = ex * w1;
    float s = 0.f, p[E_EXP];
#pragma unroll
    for (int e = 0; e < E_EXP; e++) { p[e] = __expf(lg[e] - l1); s += p[e]; }
    float inv = 1.f / (s * (float)T_TOK);
#pragma unroll
    for (int e = 0; e < E_EXP; e++) atomicAdd(&mean[e], p[e] * inv);
    int p1 = atomicAdd(&cnt[i1], 1);
    toks[i1 * T_TOK + p1] = t;  wts[i1 * T_TOK + p1] = w1;
    int p2 = atomicAdd(&cnt[i2], 1);
    toks[i2 * T_TOK + p2] = t;  wts[i2 * T_TOK + p2] = w2;
  }
}

// concat two aligned 8-half LDS loads into a WMMA fragment
__device__ __forceinline__ v16bf ld_frag(const __bf16* p0, const __bf16* p1) {
  v8bf a = *(const v8bf*)p0;
  v8bf b = *(const v8bf*)p1;
  return __builtin_shufflevector(a, b, 0, 1, 2, 3, 4, 5, 6, 7,
                                 8, 9, 10, 11, 12, 13, 14, 15);
}

// ---------------- fused expert FFN: bf16 WMMA, act tile resident in LDS ----
__global__ __launch_bounds__(256) void moe_expert(
    const __bf16* __restrict__ xbf, const float* __restrict__ w_gate,
    const float* __restrict__ w_up, const float* __restrict__ w_down,
    const int* __restrict__ cnt, const int* __restrict__ toks,
    const float* __restrict__ wts, float* __restrict__ out) {
  extern __shared__ char smem[];
  const int AP = H_DIM + 8;   // act pitch (halves): 3600 B rows (16-aligned)
  const int XP = 72;          // x staging pitch (halves): 144 B rows
  const int KP = 72;          // transposed weight pitch (halves): 144 B rows
  __bf16* act  = (__bf16*)smem;            // [64][AP]
  __bf16* xs   = act + 64 * AP;            // [64][XP]
  __bf16* wgsT = xs + 64 * XP;             // [128 n][KP]  (N-major)
  __bf16* wusT = wgsT + 128 * KP;          // [128 n][KP]
  int*    tid  = (int*)(wusT + 128 * KP);  // [64]
  float*  scw  = (float*)(tid + 64);       // [64]

  int e = blockIdx.y;
  int m_base = blockIdx.x * 64;
  int cnt_e = cnt[e];
  if (m_base >= cnt_e) return;  // uniform across block

  if (threadIdx.x < 64) {
    int r = m_base + threadIdx.x;
    tid[threadIdx.x] = (r < cnt_e) ? toks[e * T_TOK + r] : 0;
    scw[threadIdx.x] = (r < cnt_e) ? wts[e * T_TOK + r] : 0.f;
  }
  __syncthreads();

  int lane = threadIdx.x & 31;
  int wv   = threadIdx.x >> 5;   // 0..7 waves
  bool lo  = lane < 16;
  int ln   = lane & 15;
  int sr4  = threadIdx.x >> 2;   // staging row 0..63
  int sc4  = threadIdx.x & 3;    // staging col group
  int colw = wv * 16 + ln;       // this lane's B column in the 128-wide pass

  // hoisted invariants (keep out of the barrier-bracketed loops)
  const __bf16* xrow = xbf + (size_t)tid[sr4] * D_DIM + sc4 * 16;  // gathered row
  __bf16* xdst = xs + sr4 * XP + sc4 * 16;
  const size_t wg_base = (size_t)e * D_DIM * H_DIM + (size_t)sr4 * H_DIM + sc4 * 32;
  const size_t wd_base = (size_t)e * H_DIM * D_DIM + (size_t)sr4 * D_DIM + sc4 * 32;
  const v8f vzero = {0.f, 0.f, 0.f, 0.f, 0.f, 0.f, 0.f, 0.f};

  // ---------- Phase 1: act = silu(x*Wg) .* (x*Wu), 128 H-cols per pass ----
  for (int hb = 0; hb < H_DIM; hb += 128) {
    v8f accG[4], accU[4];
#pragma unroll
    for (int mt = 0; mt < 4; mt++) { accG[mt] = vzero; accU[mt] = vzero; }

    for (int kc = 0; kc < D_DIM; kc += 64) {
      { // stage x tile [64][64] bf16: pure b128 copy, no conversion
        v8bf o0 = *(const v8bf*)(xrow + kc);
        v8bf o1 = *(const v8bf*)(xrow + kc + 8);
        *(v8bf*)(xdst) = o0;
        *(v8bf*)(xdst + 8) = o1;
      }
      { // stage Wg/Wu tiles TRANSPOSED: global fp32 [k][n] -> LDS bf16 [n][k]
        const float* sg = w_gate + wg_base + (size_t)kc * H_DIM + hb;
        const float* su = w_up + wg_base + (size_t)kc * H_DIM + hb;
#pragma unroll
        for (int j = 0; j < 32; j += 4) {
          float4 vg = *(const float4*)(sg + j);
          float4 vu = *(const float4*)(su + j);
          int c = sc4 * 32 + j;
          wgsT[(c + 0) * KP + sr4] = (__bf16)vg.x;
          wgsT[(c + 1) * KP + sr4] = (__bf16)vg.y;
          wgsT[(c + 2) * KP + sr4] = (__bf16)vg.z;
          wgsT[(c + 3) * KP + sr4] = (__bf16)vg.w;
          wusT[(c + 0) * KP + sr4] = (__bf16)vu.x;
          wusT[(c + 1) * KP + sr4] = (__bf16)vu.y;
          wusT[(c + 2) * KP + sr4] = (__bf16)vu.z;
          wusT[(c + 3) * KP + sr4] = (__bf16)vu.w;
        }
        if (kc + 64 < D_DIM) {            // gfx1250 global_prefetch_b8
          __builtin_prefetch(sg + 64 * H_DIM, 0, 1);
          __builtin_prefetch(su + 64 * H_DIM, 0, 1);
        }
      }
      __syncthreads();

#pragma unroll
      for (int ks = 0; ks < 64; ks += 32) {
        const __bf16* bgp = wgsT + colw * KP + ks + (lo ? 0 : 8);
        const __bf16* bup = wusT + colw * KP + ks + (lo ? 0 : 8);
        v16bf bg = ld_frag(bgp, bgp + 16);
        v16bf bu = ld_frag(bup, bup + 16);
#pragma unroll
        for (int mt = 0; mt < 4; mt++) {
          int row = mt * 16 + ln;
          const __bf16* ap = xs + row * XP + ks + (lo ? 0 : 8);
          v16bf a = ld_frag(ap, ap + 16);   // K {0..7,16..23} / {8..15,24..31}
          accG[mt] = __builtin_amdgcn_wmma_f32_16x16x32_bf16(
              false, a, false, bg, (short)0, accG[mt], false, false);
          accU[mt] = __builtin_amdgcn_wmma_f32_16x16x32_bf16(
              false, a, false, bu, (short)0, accU[mt], false, false);
        }
      }
      __syncthreads();
    }
    // silu(G)*U -> LDS activation tile (bf16)
#pragma unroll
    for (int mt = 0; mt < 4; mt++) {
#pragma unroll
      for (int j = 0; j < 8; j++) {
        int m = mt * 16 + (lo ? j : j + 8);
        int n = hb + wv * 16 + ln;
        float g = accG[mt][j];
        float u = accU[mt][j];
        act[m * AP + n] = (__bf16)(g / (1.f + __expf(-g)) * u);
      }
    }
  }
  __syncthreads();

  // ---------- Phase 2: out += (act @ Wd) * combine, 128 D-cols per pass ----
  for (int nb = 0; nb < D_DIM; nb += 128) {
    v8f acc[4];
#pragma unroll
    for (int mt = 0; mt < 4; mt++) acc[mt] = vzero;

    for (int kc = 0; kc < H_DIM; kc += 64) {
      { // stage Wd tile TRANSPOSED into wgsT buffer
        const float* sd = w_down + wd_base + (size_t)kc * D_DIM + nb;
#pragma unroll
        for (int j = 0; j < 32; j += 4) {
          float4 v = *(const float4*)(sd + j);
          int c = sc4 * 32 + j;
          wgsT[(c + 0) * KP + sr4] = (__bf16)v.x;
          wgsT[(c + 1) * KP + sr4] = (__bf16)v.y;
          wgsT[(c + 2) * KP + sr4] = (__bf16)v.z;
          wgsT[(c + 3) * KP + sr4] = (__bf16)v.w;
        }
        if (kc + 64 < H_DIM) __builtin_prefetch(sd + 64 * D_DIM, 0, 1);
      }
      __syncthreads();

#pragma unroll
      for (int ks = 0; ks < 64; ks += 32) {
        const __bf16* bdp = wgsT + colw * KP + ks + (lo ? 0 : 8);
        v16bf bd = ld_frag(bdp, bdp + 16);
#pragma unroll
        for (int mt = 0; mt < 4; mt++) {
          int row = mt * 16 + ln;
          int kb = kc + ks;
          const __bf16* ap = act + row * AP + kb + (lo ? 0 : 8);
          v16bf a = ld_frag(ap, ap + 16);
          acc[mt] = __builtin_amdgcn_wmma_f32_16x16x32_bf16(
              false, a, false, bd, (short)0, acc[mt], false, false);
        }
      }
      __syncthreads();
    }
#pragma unroll
    for (int mt = 0; mt < 4; mt++) {
#pragma unroll
      for (int j = 0; j < 8; j++) {
        int m = mt * 16 + (lo ? j : j + 8);
        int n = nb + wv * 16 + ln;
        float v = acc[mt][j] * scw[m];
        atomicAdd(&out[(size_t)tid[m] * D_DIM + n], v);
      }
    }
  }
}

// ---------------- finish: load-balance loss + mean_probs ----------------
__global__ void moe_finish(const float* __restrict__ mean, float* __restrict__ tail) {
  if (threadIdx.x == 0) {
    float s = 0.f;
#pragma unroll
    for (int e = 0; e < E_EXP; e++) s += mean[e] * mean[e];
    tail[0] = (float)E_EXP * s;
#pragma unroll
    for (int e = 0; e < E_EXP; e++) tail[1 + e] = mean[e];
  }
}

extern "C" void kernel_launch(void* const* d_in, const int* in_sizes, int n_in,
                              void* d_out, int out_size, void* d_ws, size_t ws_size,
                              hipStream_t stream) {
  const float* x  = (const float*)d_in[0];
  const float* gw = (const float*)d_in[1];
  const float* wg = (const float*)d_in[2];
  const float* wu = (const float*)d_in[3];
  const float* wd = (const float*)d_in[4];
  float* out = (float*)d_out;

  // workspace layout (4-byte units)
  int*    cnt  = (int*)d_ws;                         // [8]
  float*  mean = (float*)d_ws + 8;                   // [8]
  int*    toks = (int*)d_ws + 64;                    // [E*T]
  float*  wts  = (float*)d_ws + 64 + E_EXP * T_TOK;  // [E*T]
  __bf16* xbf  = (__bf16*)((float*)d_ws + 64 + 2 * E_EXP * T_TOK);  // [T*D] bf16

  int initBlocks = (T_TOK * D_DIM + 255) / 256;
  moe_init<<<initBlocks, 256, 0, stream>>>(out, cnt, mean, toks, wts);
  moe_cast<<<T_TOK * D_DIM / (256 * 8), 256, 0, stream>>>(x, xbf);
  moe_gate<<<T_TOK / 8, 256, 0, stream>>>(x, gw, cnt, mean, toks, wts);

  // LDS: act 230400 + xs 9216 + 2*18432 (wT) + 512 = 276,992 B < 320 KB
  size_t smem = (size_t)(64 * (H_DIM + 8) + 64 * 72 + 2 * 128 * 72) * 2
              + 64 * 4 + 64 * 4;
  moe_expert<<<dim3(T_TOK / 64, E_EXP), 256, smem, stream>>>(
      xbf, wg, wu, wd, cnt, toks, wts, out);

  moe_finish<<<1, 32, 0, stream>>>(mean, out + (size_t)T_TOK * D_DIM);
}
